// CTMHeadLite_25340307047083
// MI455X (gfx1250) — matrix operand order repaired and checked by reading, compiled, not run on hardware
//
#include <hip/hip_runtime.h>
#include <hip/hip_bf16.h>

#define B_ROWS 16384
#define H_DIM  1024
#define F_DIM  4096
#define EPS    1e-5f

typedef __attribute__((ext_vector_type(16))) __bf16 v16bf;
typedef __attribute__((ext_vector_type(8)))  float  v8f;
typedef __attribute__((ext_vector_type(4)))  int    ivec4;
typedef __attribute__((ext_vector_type(4)))  float  fvec4;

union AFrag {
    ivec4   q[2];
    unsigned u32[8];
    v16bf   v;
};

__device__ __forceinline__ unsigned pack2(float a, float b) {
    unsigned short ua = __builtin_bit_cast(unsigned short, (__bf16)a);
    unsigned short ub = __builtin_bit_cast(unsigned short, (__bf16)b);
    return (unsigned)ua | ((unsigned)ub << 16);
}

__device__ __forceinline__ v8f wmma_bf16(v16bf a, v16bf b, v8f c) {
    return __builtin_amdgcn_wmma_f32_16x16x32_bf16(
        /*neg_a=*/false, a, /*neg_b=*/false, b,
        /*c_mod=*/(short)0, c, /*reuse_a=*/false, /*reuse_b=*/false);
}

// B-fragment (32x16 KxN tile): 512 bf16 per tile; lane-contiguous 32B each.
__device__ __forceinline__ v16bf load_b_frag(const __bf16* base, long tile, int lane) {
    const __bf16* p = base + tile * 512 + lane * 16;
    AFrag u;
    u.q[0] = *(const ivec4*)p;
    u.q[1] = *(const ivec4*)(p + 8);
    return u.v;
}

// A-fragment (16x32 bf16) from row-major bf16 matrix.
// lane<16: M=lane, K {0..7, 16..23}; lane>=16: M=lane-16, K {8..15, 24..31}.
__device__ __forceinline__ v16bf load_a_bf16(const __bf16* laneptr, int kofs) {
    AFrag u;
    u.q[0] = *(const ivec4*)(laneptr + kofs);
    u.q[1] = *(const ivec4*)(laneptr + kofs + 16);
    return u.v;
}

__device__ __forceinline__ float sigmoidf_(float x) {
    return 1.0f / (1.0f + __expf(-x));
}

// CDNA5 async copy: 16B per lane, global -> LDS, tracked by ASYNCcnt.
// LDS dest offset = low 32 bits of the generic address (ISA 10.2 LDS aperture).
__device__ __forceinline__ void async_copy_b128(void* lds_dst, const void* gsrc) {
    unsigned loff = (unsigned)(unsigned long long)lds_dst;
    asm volatile("global_load_async_to_lds_b128 %0, %1, off"
                 :: "v"(loff), "v"(gsrc) : "memory");
}
__device__ __forceinline__ void wait_async0() {
    asm volatile("s_wait_asynccnt 0x0" ::: "memory");
}

// ---------------------------------------------------------------------------
// Weight conversion: fp32 W[N][K] -> bf16 B-fragments.
// Tile t = (n/16)*(K/32) + (k/32); lane l<16 holds n=l%16, K {0..7,16..23} (2/dword);
// lane>=16 holds K {8..15,24..31}.
// ---------------------------------------------------------------------------
__global__ void cvt_weight_frags(const float* __restrict__ W,
                                 __bf16* __restrict__ dst, int N, int K) {
    int dw = blockIdx.x * 256 + threadIdx.x;
    int total = (N * K) >> 1;
    if (dw >= total) return;
    int ktiles = K >> 5;
    int tile   = dw >> 8;
    int within = dw & 255;
    int lane   = within >> 3;
    int v      = within & 7;
    int n = (tile / ktiles) * 16 + (lane & 15);
    int k = (tile % ktiles) * 32 + ((lane >> 4) << 3) + (v << 1) + ((v >> 2) << 3);
    const float* s = W + (size_t)n * K + k;
    ((unsigned*)dst)[dw] = pack2(s[0], s[1]);
}

// ---------------------------------------------------------------------------
// Input projection GEMM: concat(l0,l1)[B,2H] x W_inp^T -> y[B,H] fp32 pre-LN.
// ---------------------------------------------------------------------------
__global__ __launch_bounds__(256) void input_gemm_kernel(
    const float* __restrict__ l0, const float* __restrict__ l1,
    const __bf16* __restrict__ wfrag, const float* __restrict__ bias,
    float* __restrict__ y) {
    int wave = threadIdx.x >> 5, lane = threadIdx.x & 31;
    int tile = blockIdx.x * 8 + wave;         // rt*64 + ct
    int rt = tile >> 6, ct = tile & 63;
    int mrow = lane & 15, hi = lane >> 4;
    const float* a0 = l0 + (size_t)(rt * 16 + mrow) * H_DIM + hi * 8;
    const float* a1 = l1 + (size_t)(rt * 16 + mrow) * H_DIM + hi * 8;
    v8f acc = {};
    for (int kt = 0; kt < 64; ++kt) {
        const float* p = ((kt < 32) ? a0 : a1) + (kt & 31) * 32;
        fvec4 f0 = *(const fvec4*)p;
        fvec4 f1 = *(const fvec4*)(p + 4);
        fvec4 f2 = *(const fvec4*)(p + 16);
        fvec4 f3 = *(const fvec4*)(p + 20);
        AFrag u;
        u.u32[0] = pack2(f0[0], f0[1]); u.u32[1] = pack2(f0[2], f0[3]);
        u.u32[2] = pack2(f1[0], f1[1]); u.u32[3] = pack2(f1[2], f1[3]);
        u.u32[4] = pack2(f2[0], f2[1]); u.u32[5] = pack2(f2[2], f2[3]);
        u.u32[6] = pack2(f3[0], f3[1]); u.u32[7] = pack2(f3[2], f3[3]);
        v16bf b = load_b_frag(wfrag, (long)ct * 64 + kt, lane);
        acc = wmma_bf16(u.v, b, acc);
    }
    int col = ct * 16 + mrow;
    float bb = bias[col];
#pragma unroll
    for (int v = 0; v < 8; ++v)
        y[(size_t)(rt * 16 + v + hi * 8) * H_DIM + col] = acc[v] + bb;
}

// ---------------------------------------------------------------------------
// GRU tick, M-blocked. Block owns a 64-row panel of h, staged once in LDS via
// async-to-LDS b128 copies (padded rows to dodge bank conflicts). Each wave
// computes 32 rows x 16 columns per ct step: 12 WMMA chains share every
// B fragment across two row-subtiles (2x L2 reuse) and one LDS A fragment.
// ---------------------------------------------------------------------------
#define ROWS_PB    64
#define LDS_STRIDE 1040   // 1024 + 16 bf16 pad (32B -> 8-bank rotation per row)

__global__ __launch_bounds__(256) void gru_tick_kernel(
    const __bf16* __restrict__ h,
    const __bf16* __restrict__ wih_frag, const __bf16* __restrict__ whh_frag,
    const float* __restrict__ b_ih, const float* __restrict__ b_hh,
    __bf16* __restrict__ hnew) {
    extern __shared__ __bf16 hls[];                 // [64][LDS_STRIDE]
    int tid = threadIdx.x;
    int rb  = blockIdx.x;                           // 64-row panel index
    const __bf16* hbase = h + (size_t)rb * ROWS_PB * H_DIM;

    // Stage 64x1024 bf16 panel into LDS: 32 async b128 ops per wave (<=63 cnt).
#pragma unroll 4
    for (int i = 0; i < 32; ++i) {
        int c   = tid + 256 * i;                    // 8192 chunks of 8 elems
        int row = c >> 7;
        int k8  = (c & 127) << 3;
        async_copy_b128(&hls[row * LDS_STRIDE + k8], hbase + row * H_DIM + k8);
    }
    wait_async0();
    __syncthreads();

    int wave = tid >> 5, lane = tid & 31;
    int mrow = lane & 15, hi = lane >> 4;
    int rhalf = wave >> 2;                          // 0/1 -> rows rhalf*32..+31
    int ctg   = wave & 3;                           // cts ctg*16..+15

    const __bf16* a0p = &hls[(rhalf * 32 + mrow) * LDS_STRIDE + hi * 8];
    const __bf16* a1p = a0p + 16 * LDS_STRIDE;

    for (int cc = 0; cc < 16; ++cc) {
        int ct = ctg * 16 + cc;
        v8f acc[2][6];
#pragma unroll
        for (int i = 0; i < 2; ++i)
#pragma unroll
            for (int g = 0; g < 6; ++g) acc[i][g] = (v8f){};

        for (int kt = 0; kt < 32; ++kt) {
            v16bf a0 = load_a_bf16(a0p, kt * 32);
            v16bf a1 = load_a_bf16(a1p, kt * 32);
#pragma unroll
            for (int g = 0; g < 6; ++g) {
                const __bf16* wb = (g < 3) ? wih_frag : whh_frag;
                long nt = (long)((g % 3) * 64 + ct);
                v16bf b = load_b_frag(wb, nt * 32 + kt, lane);
                acc[0][g] = wmma_bf16(a0, b, acc[0][g]);
                acc[1][g] = wmma_bf16(a1, b, acc[1][g]);
            }
        }

        int col = ct * 16 + mrow;
        float bxr = b_ih[col], bxz = b_ih[H_DIM + col], bxn = b_ih[2 * H_DIM + col];
        float bhr = b_hh[col], bhz = b_hh[H_DIM + col], bhn = b_hh[2 * H_DIM + col];
#pragma unroll
        for (int rsub = 0; rsub < 2; ++rsub) {
#pragma unroll
            for (int v = 0; v < 8; ++v) {
                int rlocal = rhalf * 32 + rsub * 16 + v + hi * 8;
                float r = sigmoidf_(acc[rsub][0][v] + bxr + acc[rsub][3][v] + bhr);
                float z = sigmoidf_(acc[rsub][1][v] + bxz + acc[rsub][4][v] + bhz);
                float n = tanhf(acc[rsub][2][v] + bxn + r * (acc[rsub][5][v] + bhn));
                float hv = (float)hls[rlocal * LDS_STRIDE + col];
                hnew[((size_t)rb * ROWS_PB + rlocal) * H_DIM + col] =
                    (__bf16)((1.0f - z) * n + z * hv);
            }
        }
    }
}

// ---------------------------------------------------------------------------
// Feature projection GEMM: h[B,H](bf16) x W_feat^T -> y[B,F] fp32 pre-LN.
// ---------------------------------------------------------------------------
__global__ __launch_bounds__(256) void feat_gemm_kernel(
    const __bf16* __restrict__ h, const __bf16* __restrict__ wfrag,
    const float* __restrict__ bias, float* __restrict__ y) {
    int wave = threadIdx.x >> 5, lane = threadIdx.x & 31;
    int tile = blockIdx.x * 8 + wave;         // rt*256 + ct
    int rt = tile >> 8, ct = tile & 255;
    int mrow = lane & 15, hi = lane >> 4;
    const __bf16* aptr = h + (size_t)(rt * 16 + mrow) * H_DIM + hi * 8;
    v8f acc = {};
    for (int kt = 0; kt < 32; ++kt) {
        v16bf a = load_a_bf16(aptr, kt * 32);
        v16bf b = load_b_frag(wfrag, (long)ct * 32 + kt, lane);
        acc = wmma_bf16(a, b, acc);
    }
    int col = ct * 16 + mrow;
    float bb = bias[col];
#pragma unroll
    for (int v = 0; v < 8; ++v)
        y[(size_t)(rt * 16 + v + hi * 8) * F_DIM + col] = acc[v] + bb;
}

// ---------------------------------------------------------------------------
// Per-row LayerNorm; writes fp32 (in-place safe) or bf16.
// ---------------------------------------------------------------------------
__global__ void rownorm_kernel(const float* __restrict__ y, int ncols,
                               const float* __restrict__ g,
                               const float* __restrict__ be,
                               float* outf, __bf16* outb) {
    extern __shared__ float sred[];
    int row = blockIdx.x, tid = threadIdx.x, nt = blockDim.x;
    const float* yp = y + (size_t)row * ncols;
    float s = 0.0f, ss = 0.0f;
    for (int c = tid; c < ncols; c += nt) { float x = yp[c]; s += x; ss += x * x; }
    sred[tid] = s; sred[nt + tid] = ss;
    __syncthreads();
    if (tid == 0) {
        float S = 0.0f, SS = 0.0f;
        for (int i = 0; i < nt; ++i) { S += sred[i]; SS += sred[nt + i]; }
        float mu = S / (float)ncols;
        float var = SS / (float)ncols - mu * mu;
        sred[2 * nt] = mu;
        sred[2 * nt + 1] = rsqrtf(var + EPS);
    }
    __syncthreads();
    float mu = sred[2 * nt], rs = sred[2 * nt + 1];
    for (int c = tid; c < ncols; c += nt) {
        float xn = (yp[c] - mu) * rs * g[c] + be[c];
        if (outb) outb[(size_t)row * ncols + c] = (__bf16)xn;
        else      outf[(size_t)row * ncols + c] = xn;
    }
}

// ---------------------------------------------------------------------------
// Head: synchronization = mean(h,-1); certainty = sigmoid(h . W_cert + b).
// ---------------------------------------------------------------------------
__global__ __launch_bounds__(256) void head_kernel(
    const __bf16* __restrict__ h, const float* __restrict__ wc,
    const float* __restrict__ bc, float* __restrict__ sync_out,
    float* __restrict__ cert_out) {
    int wave = threadIdx.x >> 5, lane = threadIdx.x & 31;
    int row = blockIdx.x * 8 + wave;
    const __bf16* hp = h + (size_t)row * H_DIM;
    float s = 0.0f, d = 0.0f;
    for (int k = lane; k < H_DIM; k += 32) {
        float x = (float)hp[k];
        s += x; d += x * wc[k];
    }
#pragma unroll
    for (int o = 16; o > 0; o >>= 1) {
        s += __shfl_down(s, o, 32);
        d += __shfl_down(d, o, 32);
    }
    if (lane == 0) {
        sync_out[row] = s * (1.0f / (float)H_DIM);
        cert_out[row] = sigmoidf_(d + bc[0]);
    }
}

// ---------------------------------------------------------------------------
extern "C" void kernel_launch(void* const* d_in, const int* in_sizes, int n_in,
                              void* d_out, int out_size, void* d_ws, size_t ws_size,
                              hipStream_t stream) {
    (void)in_sizes; (void)n_in; (void)out_size; (void)ws_size;
    const float* l0     = (const float*)d_in[0];
    const float* l1     = (const float*)d_in[1];
    const float* W_inp  = (const float*)d_in[2];
    const float* b_inp  = (const float*)d_in[3];
    const float* g1     = (const float*)d_in[4];
    const float* be1    = (const float*)d_in[5];
    const float* W_ih   = (const float*)d_in[6];
    const float* b_ih   = (const float*)d_in[7];
    const float* W_hh   = (const float*)d_in[8];
    const float* b_hh   = (const float*)d_in[9];
    const float* W_feat = (const float*)d_in[10];
    const float* b_feat = (const float*)d_in[11];
    const float* g2     = (const float*)d_in[12];
    const float* be2    = (const float*)d_in[13];
    const float* W_cert = (const float*)d_in[14];
    const float* b_cert = (const float*)d_in[15];

    // Workspace: bf16 weight fragments + double-buffered h (~88 MB).
    char* ws = (char*)d_ws;
    size_t off = 0;
    __bf16* fr_inp  = (__bf16*)(ws + off); off += (size_t)1024 * 2048 * 2;
    __bf16* fr_ih   = (__bf16*)(ws + off); off += (size_t)3072 * 1024 * 2;
    __bf16* fr_hh   = (__bf16*)(ws + off); off += (size_t)3072 * 1024 * 2;
    __bf16* fr_feat = (__bf16*)(ws + off); off += (size_t)4096 * 1024 * 2;
    __bf16* h_a     = (__bf16*)(ws + off); off += (size_t)B_ROWS * H_DIM * 2;
    __bf16* h_b     = (__bf16*)(ws + off);

    float* features = (float*)d_out;                          // [B, F]
    float* syncv    = features + (size_t)B_ROWS * F_DIM;      // [B]
    float* certv    = syncv + B_ROWS;                         // [B]

    // 1) Weight conversion to WMMA B-fragments (bf16).
    cvt_weight_frags<<<(1024 * 2048 / 2 + 255) / 256, 256, 0, stream>>>(W_inp, fr_inp, 1024, 2048);
    cvt_weight_frags<<<(3072 * 1024 / 2 + 255) / 256, 256, 0, stream>>>(W_ih, fr_ih, 3072, 1024);
    cvt_weight_frags<<<(3072 * 1024 / 2 + 255) / 256, 256, 0, stream>>>(W_hh, fr_hh, 3072, 1024);
    cvt_weight_frags<<<(4096 * 1024 / 2 + 255) / 256, 256, 0, stream>>>(W_feat, fr_feat, 4096, 1024);

    // 2) Input projection -> pre-LN y (scratch in features region), LN -> h0.
    input_gemm_kernel<<<(B_ROWS / 16) * (H_DIM / 16) / 8, 256, 0, stream>>>(
        l0, l1, fr_inp, b_inp, features);
    rownorm_kernel<<<B_ROWS, 256, (2 * 256 + 2) * sizeof(float), stream>>>(
        features, H_DIM, g1, be1, nullptr, h_a);

    // 3) 5 GRU ticks (ping-pong h buffers), LDS-staged 64-row panels.
    __bf16* hc = h_a;
    __bf16* hn = h_b;
    for (int t = 0; t < 5; ++t) {
        gru_tick_kernel<<<B_ROWS / ROWS_PB, 256,
                          ROWS_PB * LDS_STRIDE * sizeof(__bf16), stream>>>(
            hc, fr_ih, fr_hh, b_ih, b_hh, hn);
        __bf16* tmp = hc; hc = hn; hn = tmp;
    }

    // 4) Feature projection -> d_out, then in-place LayerNorm over F_DIM.
    feat_gemm_kernel<<<(B_ROWS / 16) * (F_DIM / 16) / 8, 256, 0, stream>>>(
        hc, fr_feat, b_feat, features);
    rownorm_kernel<<<B_ROWS, 256, (2 * 256 + 2) * sizeof(float), stream>>>(
        features, F_DIM, g2, be2, features, nullptr);

    // 5) Synchronization mean + certainty head.
    head_kernel<<<B_ROWS / 8, 256, 0, stream>>>(hc, W_cert, b_cert, syncv, certv);
}
